// GlobalDownSample_85676007620987
// MI455X (gfx1250) — matrix operand order repaired and checked by reading, not compiled, not run on hardware
//
#include <hip/hip_runtime.h>
#include <hip/hip_bf16.h>

#define BB 8
#define CC 128
#define NN 2048
#define MM 512

typedef __attribute__((ext_vector_type(16))) __bf16 v16bf;
typedef __attribute__((ext_vector_type(8)))  float  v8f;

union ABfrag {
    v16bf    v;
    unsigned u[8];
    __bf16   h[16];
    uint4    q4[2];
};

union DStore {
    uint4  q;
    __bf16 h[8];
};

union AccU {
    v8f    v;
    float4 f4[2];
    float  f[8];
};

__device__ __forceinline__ __bf16 f2bf(float f) {
    unsigned u = __builtin_bit_cast(unsigned, f);
    unsigned r = (u + 0x7FFFu + ((u >> 16) & 1u)) >> 16;
    unsigned short hs = (unsigned short)r;
    return __builtin_bit_cast(__bf16, hs);
}

// A-fragment (16x32 bf16 MxK): lane row = lane&15, half = lane>>4.
// VGPR v<4 holds K = 8*half+2v,+1 ; v>=4 holds K = 16+8*half+2(v-4),+1.
// => two contiguous 8-element runs: [8*half .. +7] and [16+8*half .. +7].
__device__ __forceinline__ void load_a(ABfrag& f, const __bf16* rowbase,
                                       int ks, int half) {
    f.q4[0] = *(const uint4*)(rowbase + ks + 8 * half);
    f.q4[1] = *(const uint4*)(rowbase + ks + 16 + 8 * half);
}
// B-fragment (32x16 bf16 KxN): lane col = lane&15, half = lane>>4.
// VGPR v<4 holds K = 8*half+2v,+1 ; v>=4 holds K = 16+8*half+2(v-4),+1.
__device__ __forceinline__ void load_b(ABfrag& f, const __bf16* rowbase,
                                       int ks, int half) {
    f.q4[0] = *(const uint4*)(rowbase + ks + 8 * half);
    f.q4[1] = *(const uint4*)(rowbase + ks + 16 + 8 * half);
}

// ---------------- K0a: plain f32 -> bf16 (weights) ----------------
__global__ void cvt_bf16_kernel(const float* __restrict__ src,
                                __bf16* __restrict__ dst, int n) {
    int i = blockIdx.x * blockDim.x + threadIdx.x;
    if (i < n) dst[i] = f2bf(src[i]);
}

// ---------------- K0b: x [B,C,N] f32 -> xT [B,N,C] bf16 (LDS tiled) -------
__global__ void cvt_transpose_kernel(const float* __restrict__ X,
                                     __bf16* __restrict__ XT) {
    __shared__ __bf16 t[32][33];
    int n0 = blockIdx.x * 32, c0 = blockIdx.y * 32, b = blockIdx.z;
    int tx = threadIdx.x, ty = threadIdx.y;
#pragma unroll
    for (int i = 0; i < 4; ++i) {
        int c = c0 + ty + i * 8;
        t[ty + i * 8][tx] = f2bf(X[((size_t)b * CC + c) * NN + n0 + tx]);
    }
    __syncthreads();
#pragma unroll
    for (int i = 0; i < 4; ++i) {
        int n = n0 + ty + i * 8;
        XT[((size_t)b * NN + n) * CC + c0 + tx] = t[tx][ty + i * 8];
    }
}

// ---------------- K1: Y[o,n] = sum_c W[o,c] * xT[n,c] ----------------
// one wave per 16x16 tile; store_t: Y is [B,N,C] (q/k), else [B,C,N] (v)
__global__ void qkv_gemm_kernel(const __bf16* __restrict__ W,
                                const __bf16* __restrict__ XT,
                                __bf16* __restrict__ Y, int store_t) {
    int wavei = (blockIdx.x * blockDim.x + threadIdx.x) >> 5;
    int lane  = threadIdx.x & 31;
    const int ntn = NN / 16, nto = CC / 16;
    int b   = wavei / (nto * ntn);
    int rem = wavei % (nto * ntn);
    int o0  = (rem / ntn) * 16;
    int n0  = (rem % ntn) * 16;
    int col = lane & 15, half = lane >> 4;

    const __bf16* Wr = W + (size_t)(o0 + col) * CC;
    const __bf16* Xr = XT + ((size_t)b * NN + n0 + col) * CC;

    v8f acc = {};
#pragma unroll
    for (int ks = 0; ks < CC; ks += 32) {
        ABfrag a, bf;
        load_a(a, Wr, ks, half);
        load_b(bf, Xr, ks, half);
        acc = __builtin_amdgcn_wmma_f32_16x16x32_bf16(
            false, a.v, false, bf.v, (short)0, acc, false, false);
    }
    if (store_t) {
        DStore st;
#pragma unroll
        for (int r = 0; r < 8; ++r) st.h[r] = f2bf(acc[r]);
        *(uint4*)(Y + ((size_t)b * NN + n0 + col) * CC + o0 + 8 * half) = st.q;
    } else {
        __bf16* Yb = Y + (size_t)b * CC * NN;
#pragma unroll
        for (int r = 0; r < 8; ++r)
            Yb[(size_t)(o0 + 8 * half + r) * NN + n0 + col] = f2bf(acc[r]);
    }
}

// ---------------- K2: fused energy + softmax stats + column sums ----------
// block = 4 waves; 16-row energy block [16][NN] f32 in dynamic LDS
__global__ void energy_softmax_kernel(const __bf16* __restrict__ QT,
                                      const __bf16* __restrict__ KT,
                                      float* __restrict__ rowmax,
                                      float* __restrict__ rowsum,
                                      float* __restrict__ sel) {
    extern __shared__ char smem[];
    float* esh = (float*)smem;  // [16][NN]
    __shared__ float red[16 * 8];
    __shared__ float rmx[16];
    __shared__ float rsm[16];

    const float inv = 0.088388347648318447f;  // 1/sqrt(128)
    int b    = blockIdx.x / (NN / 16);
    int n0   = (blockIdx.x % (NN / 16)) * 16;
    int wave = threadIdx.x >> 5;
    int lane = threadIdx.x & 31;
    int col  = lane & 15, half = lane >> 4;

    const __bf16* Qr = QT + ((size_t)b * NN + n0 + col) * CC;

    ABfrag afrag[4];
#pragma unroll
    for (int s = 0; s < 4; ++s) load_a(afrag[s], Qr, s * 32, half);

    for (int mt = wave; mt < NN / 16; mt += 4) {
        int m0 = mt * 16;
        const __bf16* Kr = KT + ((size_t)b * NN + m0 + col) * CC;
        __builtin_prefetch(KT + ((size_t)b * NN + m0 + 64 + col) * CC, 0, 1);
        v8f acc = {};
#pragma unroll
        for (int s = 0; s < 4; ++s) {
            ABfrag bf;
            load_b(bf, Kr, s * 32, half);
            acc = __builtin_amdgcn_wmma_f32_16x16x32_bf16(
                false, afrag[s].v, false, bf.v, (short)0, acc, false, false);
        }
#pragma unroll
        for (int r = 0; r < 8; ++r)
            esh[(8 * half + r) * NN + m0 + col] = acc[r] * inv;
    }
    __syncthreads();

    {   // per-row max (8 threads per row)
        int row = threadIdx.x >> 3, seg = threadIdx.x & 7;
        float mx = -3.0e38f;
        for (int j = seg * 256; j < (seg + 1) * 256; ++j)
            mx = fmaxf(mx, esh[row * NN + j]);
        red[row * 8 + seg] = mx;
    }
    __syncthreads();
    if (threadIdx.x < 16) {
        float mx = red[threadIdx.x * 8];
        for (int j = 1; j < 8; ++j) mx = fmaxf(mx, red[threadIdx.x * 8 + j]);
        rmx[threadIdx.x] = mx;
    }
    __syncthreads();
    {   // per-row exp-sum
        int row = threadIdx.x >> 3, seg = threadIdx.x & 7;
        float mx = rmx[row], s = 0.f;
        for (int j = seg * 256; j < (seg + 1) * 256; ++j)
            s += __expf(esh[row * NN + j] - mx);
        red[row * 8 + seg] = s;
    }
    __syncthreads();
    if (threadIdx.x < 16) {
        float s = 0.f;
        for (int j = 0; j < 8; ++j) s += red[threadIdx.x * 8 + j];
        rsm[threadIdx.x] = s;
        rowmax[b * NN + n0 + threadIdx.x] = rmx[threadIdx.x];
        rowsum[b * NN + n0 + threadIdx.x] = s;
    }
    __syncthreads();
    for (int m = (int)threadIdx.x; m < NN; m += blockDim.x) {
        float a = 0.f;
#pragma unroll
        for (int r = 0; r < 16; ++r)
            a += __expf(esh[r * NN + m] - rmx[r]) / rsm[r];
        atomicAdd(&sel[b * NN + m], a);
    }
}

// ---------------- K3: top-512 by stable rank, one block per batch ---------
__global__ void topk_kernel(const float* __restrict__ sel,
                            int* __restrict__ idx) {
    __shared__ float s[NN];
    int b = blockIdx.x;
    for (int i = threadIdx.x; i < NN; i += blockDim.x) s[i] = sel[b * NN + i];
    __syncthreads();
    for (int i = threadIdx.x; i < NN; i += blockDim.x) {
        float vi = s[i];
        int rank = 0;
        for (int j = 0; j < NN; ++j) {
            float vj = s[j];
            rank += (vj > vi) || (vj == vi && j < i);
        }
        if (rank < MM) idx[b * MM + rank] = i;
    }
}

// ---------------- K4: recompute selected attention rows + scores @ v^T ----
__global__ void out_gemm_kernel(const __bf16* __restrict__ QT,
                                const __bf16* __restrict__ KT,
                                const __bf16* __restrict__ V,
                                const float* __restrict__ rowmax,
                                const float* __restrict__ rowsum,
                                const int* __restrict__ idx,
                                float* __restrict__ out) {
    extern __shared__ char smem[];
    __bf16* att = (__bf16*)smem;  // [16][NN]
    __shared__ int   ridx[16];
    __shared__ float rmx[16];
    __shared__ float rsm[16];

    const float inv = 0.088388347648318447f;
    int b    = blockIdx.x / (MM / 16);
    int m0   = (blockIdx.x % (MM / 16)) * 16;
    int wave = threadIdx.x >> 5;
    int lane = threadIdx.x & 31;
    int col  = lane & 15, half = lane >> 4;

    if (threadIdx.x < 16) {
        int r = idx[b * MM + m0 + threadIdx.x];
        ridx[threadIdx.x] = r;
        rmx[threadIdx.x]  = rowmax[b * NN + r];
        rsm[threadIdx.x]  = rowsum[b * NN + r];
    }
    __syncthreads();

    // A = q rows (transposed layout) at the 16 selected indices
    ABfrag afrag[4];
    {
        const __bf16* Qr = QT + ((size_t)b * NN + ridx[col]) * CC;
#pragma unroll
        for (int s = 0; s < 4; ++s) load_a(afrag[s], Qr, s * 32, half);
    }

    for (int nt = wave; nt < NN / 16; nt += 4) {
        int n0 = nt * 16;
        const __bf16* Kr = KT + ((size_t)b * NN + n0 + col) * CC;
        __builtin_prefetch(KT + ((size_t)b * NN + n0 + 64 + col) * CC, 0, 1);
        v8f acc = {};
#pragma unroll
        for (int s = 0; s < 4; ++s) {
            ABfrag bf;
            load_b(bf, Kr, s * 32, half);
            acc = __builtin_amdgcn_wmma_f32_16x16x32_bf16(
                false, afrag[s].v, false, bf.v, (short)0, acc, false, false);
        }
#pragma unroll
        for (int r = 0; r < 8; ++r) {
            int row = 8 * half + r;
            float a = __expf(acc[r] * inv - rmx[row]) / rsm[row];
            att[row * NN + n0 + col] = f2bf(a);
        }
    }
    __syncthreads();

    // out tile [16 m x 16 c] = att[16 x NN] @ v^T[NN x 16]
    for (int ct = wave; ct < CC / 16; ct += 4) {
        int c0 = ct * 16;
        const __bf16* Ar = att + (size_t)col * NN;
        const __bf16* Vr = V + ((size_t)b * CC + c0 + col) * NN;
        v8f acc = {};
        for (int ks = 0; ks < NN; ks += 32) {
            ABfrag a2, b2;
            load_a(a2, Ar, ks, half);
            load_b(b2, Vr, ks, half);
            acc = __builtin_amdgcn_wmma_f32_16x16x32_bf16(
                false, a2.v, false, b2.v, (short)0, acc, false, false);
        }
        AccU au;
        au.v = acc;
        float* obase = out + ((size_t)b * CC + c0 + col) * MM + m0 + 8 * half;
        *(float4*)(obase)     = au.f4[0];
        *(float4*)(obase + 4) = au.f4[1];
    }
}

extern "C" void kernel_launch(void* const* d_in, const int* in_sizes, int n_in,
                              void* d_out, int out_size, void* d_ws, size_t ws_size,
                              hipStream_t stream) {
    const float* x  = (const float*)d_in[0];
    const float* Wq = (const float*)d_in[1];
    const float* Wk = (const float*)d_in[2];
    const float* Wv = (const float*)d_in[3];
    float* out = (float*)d_out;

    char* ws = (char*)d_ws;
    size_t off = 0;
    auto alloc = [&](size_t bytes) -> char* {
        char* p = ws + off;
        off += (bytes + 255) & ~(size_t)255;
        return p;
    };
    __bf16* xT  = (__bf16*)alloc((size_t)BB * NN * CC * 2);  // [B,N,C]
    __bf16* wqb = (__bf16*)alloc((size_t)CC * CC * 2);
    __bf16* wkb = (__bf16*)alloc((size_t)CC * CC * 2);
    __bf16* wvb = (__bf16*)alloc((size_t)CC * CC * 2);
    __bf16* qT  = (__bf16*)alloc((size_t)BB * NN * CC * 2);  // [B,N,C]
    __bf16* kT  = (__bf16*)alloc((size_t)BB * NN * CC * 2);  // [B,N,C]
    __bf16* vb  = (__bf16*)alloc((size_t)BB * CC * NN * 2);  // [B,C,N]
    float*  rmaxp = (float*)alloc((size_t)BB * NN * 4);
    float*  rsump = (float*)alloc((size_t)BB * NN * 4);
    float*  selp  = (float*)alloc((size_t)BB * NN * 4);
    int*    idxp  = (int*)alloc((size_t)BB * MM * 4);

    cvt_transpose_kernel<<<dim3(NN / 32, CC / 32, BB), dim3(32, 8), 0, stream>>>(x, xT);
    cvt_bf16_kernel<<<(CC * CC + 255) / 256, 256, 0, stream>>>(Wq, wqb, CC * CC);
    cvt_bf16_kernel<<<(CC * CC + 255) / 256, 256, 0, stream>>>(Wk, wkb, CC * CC);
    cvt_bf16_kernel<<<(CC * CC + 255) / 256, 256, 0, stream>>>(Wv, wvb, CC * CC);
    hipMemsetAsync(selp, 0, (size_t)BB * NN * sizeof(float), stream);

    const int tiles = BB * (CC / 16) * (NN / 16);  // 8192 waves, 8 waves/block
    qkv_gemm_kernel<<<tiles / 8, 256, 0, stream>>>(wqb, xT, qT, 1);
    qkv_gemm_kernel<<<tiles / 8, 256, 0, stream>>>(wkb, xT, kT, 1);
    qkv_gemm_kernel<<<tiles / 8, 256, 0, stream>>>(wvb, xT, vb, 0);

    energy_softmax_kernel<<<BB * (NN / 16), 128, 16 * NN * sizeof(float), stream>>>(
        qT, kT, rmaxp, rsump, selp);

    topk_kernel<<<BB, 256, 0, stream>>>(selp, idxp);

    out_gemm_kernel<<<BB * (MM / 16), 128, 16 * NN * sizeof(__bf16), stream>>>(
        qT, kT, vb, rmaxp, rsump, idxp, out);
}
